// EGNN_Layer_2216203125286
// MI455X (gfx1250) — compile-verified, matrix operand order checked by reading, not run on hardware
//
#include <hip/hip_runtime.h>

#define DEV __device__ __forceinline__

typedef __attribute__((ext_vector_type(16))) _Float16 v16h;
typedef __attribute__((ext_vector_type(8)))  _Float16 v8h;
typedef __attribute__((ext_vector_type(8)))  float    v8f;
typedef __attribute__((ext_vector_type(4)))  float    v4f;

static constexpr int NN  = 50000;
static constexpr int EE  = 800000;
static constexpr int HH  = 128;
static constexpr int CAP = 64;     // per-node edge-list capacity (mean degree 16)

DEV v8f wmma16(v16h a, v16h b, v8f c) {
    return __builtin_amdgcn_wmma_f32_16x16x32_f16(false, a, false, b, (short)0, c, false, false);
}
DEV float silu_f(float v) { return v / (1.0f + __expf(-v)); }

union V16U { v16h v; v8h h[2]; };
DEV v16h ld_b2(const _Float16* p0, const _Float16* p1) {
    V16U u; u.h[0] = *(const v8h*)p0; u.h[1] = *(const v8h*)p1; return u.v;
}
DEV v16h ld_frag(const _Float16* p) { return *(const v16h*)p; }

// Pack W[K,Nout] (row-major fp32) into WMMA A-fragments of W^T.
// Fragment (mt,kk): lane L holds m = mt*16+L%16, 16 f16 along K with the
// {0..7,16..23}/{8..15,24..31} lane split. 1 KB per fragment (32 lanes x 32B).
__global__ __launch_bounds__(256) void pack_wT(const float* __restrict__ W,
                                               _Float16* __restrict__ out,
                                               int KK, int Nout, int total) {
    int tid = blockIdx.x * blockDim.x + threadIdx.x;
    if (tid >= total) return;
    int lane = tid & 31, frag = tid >> 5;
    int kk = frag % KK, mt = frag / KK;
    int m   = mt * 16 + (lane & 15);
    int off = (lane & 16) ? 8 : 0;
    _Float16* dst = out + (size_t)frag * 512 + lane * 16;
#pragma unroll
    for (int i = 0; i < 16; ++i) {
        int k = kk * 32 + off + (i < 8 ? i : i + 8);
        dst[i] = (_Float16)W[(size_t)k * Nout + m];
    }
}

__global__ __launch_bounds__(256) void init_k(const float* __restrict__ h,
                                              _Float16* __restrict__ h16,
                                              int* __restrict__ cnt) {
    int i = blockIdx.x * blockDim.x + threadIdx.x;
    if (i < NN * HH) h16[i] = (_Float16)h[i];
    if (i < NN)      cnt[i] = 0;
}

// Bucket scatter: per-node edge lists via one int atomic per edge.
__global__ __launch_bounds__(256) void scatter_k(const long long* __restrict__ ei,
                                                 int* __restrict__ cnt,
                                                 int* __restrict__ eids) {
    int e = blockIdx.x * blockDim.x + threadIdx.x;
    if (e >= EE) return;
    int n = (int)ei[e];
    int pos = atomicAdd(&cnt[n], 1);
    if (pos < CAP) eids[n * CAP + pos] = e;
}

// One wave = 16 edges (edges = B columns, out-features = A rows).
// Weight A-fragments staged in LDS once per block (128 KB of the 320 KB WGP LDS).
// Chain GEMM1(K=256,+rank-1 radial) -> GEMM2(K=128) -> coordGEMM(K=128) fully
// in registers; store m_ij (f16) and force vectors -- no fp32 atomics.
__global__ __launch_bounds__(256) void egnn_edge_kernel(
    const _Float16* __restrict__ h16, const float* __restrict__ x,
    const long long* __restrict__ ei,
    const _Float16* __restrict__ wedge,   // packed We1(64K) | We2(32K) | Wc1(32K)
    const float* __restrict__ We1last, const float* __restrict__ be1,
    const float* __restrict__ be2, const float* __restrict__ bc1,
    const float* __restrict__ Wc2,
    _Float16* __restrict__ m_e, float* __restrict__ fv) {
    extern __shared__ char smem[];
    // Stage all edge-path weight fragments into LDS (flat 128 KB copy).
    for (int c = threadIdx.x; c < (131072 / 32); c += 256)
        ((v16h*)smem)[c] = ((const v16h*)wedge)[c];
    __syncthreads();
    const _Float16* lWe1 = (const _Float16*)(smem);
    const _Float16* lWe2 = (const _Float16*)(smem + 65536);
    const _Float16* lWc1 = (const _Float16*)(smem + 98304);

    const int lane = threadIdx.x & 31;
    const int wave = blockIdx.x * (blockDim.x >> 5) + (threadIdx.x >> 5);
    const int e0 = wave << 4;
    if (e0 >= EE) return;
    const int e = e0 + (lane & 15);
    const long long row = ei[e];
    const long long col = ei[EE + e];
    const int off = (lane & 16) ? 8 : 0;

    const float dx = x[row * 3 + 0] - x[col * 3 + 0];
    const float dy = x[row * 3 + 1] - x[col * 3 + 1];
    const float dz = x[row * 3 + 2] - x[col * 3 + 2];
    const float r2 = dx * dx + dy * dy + dz * dz;
    const float dist = sqrtf(r2);

    // ---- GEMM1: t1^T = silu(We1^T @ edge_input^T + be1), radial rank-1 folded in
    v8f acc[8];
#pragma unroll
    for (int t = 0; t < 8; ++t) {
        const float* bp = be1 + t * 16 + off;
        const float* wp = We1last + t * 16 + off;
        v4f b0 = *(const v4f*)bp, b1 = *(const v4f*)(bp + 4);
        v4f w0 = *(const v4f*)wp, w1 = *(const v4f*)(wp + 4);
#pragma unroll
        for (int j = 0; j < 4; ++j) {
            acc[t][j]     = b0[j] + w0[j] * r2;
            acc[t][4 + j] = b1[j] + w1[j] * r2;
        }
    }
    const _Float16* hr = h16 + (size_t)row * HH;
    const _Float16* hc = h16 + (size_t)col * HH;
#pragma unroll
    for (int kk = 0; kk < 8; ++kk) {
        const _Float16* src = (kk < 4 ? hr : hc) + (kk & 3) * 32 + off;
        v16h b = ld_b2(src, src + 16);
#pragma unroll
        for (int t = 0; t < 8; ++t) {
            v16h a = ld_frag(lWe1 + (size_t)(t * 8 + kk) * 512 + lane * 16);
            acc[t] = wmma16(a, b, acc[t]);
        }
    }
#pragma unroll
    for (int t = 0; t < 8; ++t)
#pragma unroll
        for (int j = 0; j < 8; ++j) acc[t][j] = silu_f(acc[t][j]);

    // D -> B fragments in registers (lane=edge already matches B columns)
    v16h bf[4];
#pragma unroll
    for (int kk = 0; kk < 4; ++kk)
#pragma unroll
        for (int i = 0; i < 8; ++i) {
            bf[kk][i]     = (_Float16)acc[2 * kk][i];
            bf[kk][i + 8] = (_Float16)acc[2 * kk + 1][i];
        }

    // ---- GEMM2: m_ij^T = silu(We2^T @ t1^T + be2)
    v8f m2[8];
#pragma unroll
    for (int t = 0; t < 8; ++t) {
        const float* bp = be2 + t * 16 + off;
        v4f b0 = *(const v4f*)bp, b1 = *(const v4f*)(bp + 4);
#pragma unroll
        for (int j = 0; j < 4; ++j) { m2[t][j] = b0[j]; m2[t][4 + j] = b1[j]; }
    }
#pragma unroll
    for (int kk = 0; kk < 4; ++kk)
#pragma unroll
        for (int t = 0; t < 8; ++t) {
            v16h a = ld_frag(lWe2 + (size_t)(t * 4 + kk) * 512 + lane * 16);
            m2[t] = wmma16(a, bf[kk], m2[t]);
        }
#pragma unroll
    for (int t = 0; t < 8; ++t)
#pragma unroll
        for (int j = 0; j < 8; ++j) m2[t][j] = silu_f(m2[t][j]);

    // Store m_ij as f16 [E,128] for the node-side gather (each element read once).
    _Float16* mp = m_e + (size_t)e * HH + off;
#pragma unroll
    for (int t = 0; t < 8; ++t) {
        v8h st;
#pragma unroll
        for (int j = 0; j < 8; ++j) st[j] = (_Float16)m2[t][j];
        *(v8h*)(mp + t * 16) = st;
    }

    // ---- coord MLP: t3^T = silu(Wc1^T @ m_ij^T + bc1); force = tanh(Wc2.t3)*0.1
    v16h cf[4];
#pragma unroll
    for (int kk = 0; kk < 4; ++kk)
#pragma unroll
        for (int i = 0; i < 8; ++i) {
            cf[kk][i]     = (_Float16)m2[2 * kk][i];
            cf[kk][i + 8] = (_Float16)m2[2 * kk + 1][i];
        }
    v8f t3[8];
#pragma unroll
    for (int t = 0; t < 8; ++t) {
        const float* bp = bc1 + t * 16 + off;
        v4f b0 = *(const v4f*)bp, b1 = *(const v4f*)(bp + 4);
#pragma unroll
        for (int j = 0; j < 4; ++j) { t3[t][j] = b0[j]; t3[t][4 + j] = b1[j]; }
    }
#pragma unroll
    for (int kk = 0; kk < 4; ++kk)
#pragma unroll
        for (int t = 0; t < 8; ++t) {
            v16h a = ld_frag(lWc1 + (size_t)(t * 4 + kk) * 512 + lane * 16);
            t3[t] = wmma16(a, cf[kk], t3[t]);
        }
    float s = 0.0f;
#pragma unroll
    for (int t = 0; t < 8; ++t) {
        const float* wp = Wc2 + t * 16 + off;
        v4f w0 = *(const v4f*)wp, w1 = *(const v4f*)(wp + 4);
#pragma unroll
        for (int j = 0; j < 4; ++j) {
            s += silu_f(t3[t][j]) * w0[j];
            s += silu_f(t3[t][4 + j]) * w1[j];
        }
    }
    s += __shfl_xor(s, 16, 32);
    const float force = tanhf(s) * 0.1f;

    if ((lane & 16) == 0) {               // one lane per edge stores its force vec
        const float inv = force / (dist + 1e-8f);
        fv[(size_t)e * 3 + 0] = dx * inv;
        fv[(size_t)e * 3 + 1] = dy * inv;
        fv[(size_t)e * 3 + 2] = dz * inv;
    }
}

// One wave = 16 nodes: gather-sum m_ij + forces from edge lists (registers),
// node MLP (WMMA, weights in LDS), residual + LayerNorm, write h_new and x_new.
__global__ __launch_bounds__(256) void egnn_node_kernel(
    const float* __restrict__ h, const _Float16* __restrict__ h16,
    const float* __restrict__ x,
    const int* __restrict__ cnt, const int* __restrict__ eids,
    const _Float16* __restrict__ m_e, const float* __restrict__ fv,
    const _Float16* __restrict__ wnode,   // packed Wn1(64K) | Wn2(32K)
    const float* __restrict__ bn1, const float* __restrict__ bn2,
    const float* __restrict__ lng, const float* __restrict__ lnb,
    float* __restrict__ outh, float* __restrict__ outx) {
    extern __shared__ char smem[];
    for (int c = threadIdx.x; c < (98304 / 32); c += 256)
        ((v16h*)smem)[c] = ((const v16h*)wnode)[c];
    __syncthreads();
    const _Float16* lWn1 = (const _Float16*)(smem);
    const _Float16* lWn2 = (const _Float16*)(smem + 65536);

    const int lane = threadIdx.x & 31;
    const int wave = blockIdx.x * (blockDim.x >> 5) + (threadIdx.x >> 5);
    const int n0 = wave << 4;
    if (n0 >= NN) return;
    const int n = n0 + (lane & 15);
    const int off = (lane & 16) ? 8 : 0;

    // ---- gather-sum this node's edges: m_i (64 feats/lane) + force sum ----
    v8f mi_[8];
#pragma unroll
    for (int t = 0; t < 8; ++t)
#pragma unroll
        for (int j = 0; j < 8; ++j) mi_[t][j] = 0.0f;
    float fx = 0.0f, fy = 0.0f, fz = 0.0f;
    const int nc = min(cnt[n], CAP);
    const int* el = eids + n * CAP;
    for (int it = 0; it < nc; ++it) {
        const int ed = el[it];
        const _Float16* mp = m_e + (size_t)ed * HH + off;
#pragma unroll
        for (int t = 0; t < 8; ++t) {
            v8h m8 = *(const v8h*)(mp + t * 16);
#pragma unroll
            for (int j = 0; j < 8; ++j) mi_[t][j] += (float)m8[j];
        }
        if ((lane & 16) == 0) {
            fx += fv[(size_t)ed * 3 + 0];
            fy += fv[(size_t)ed * 3 + 1];
            fz += fv[(size_t)ed * 3 + 2];
        }
    }

    // ---- GEMM n1: K=256 = [h | m_i] ----
    v8f acc[8];
#pragma unroll
    for (int t = 0; t < 8; ++t) {
        const float* bp = bn1 + t * 16 + off;
        v4f b0 = *(const v4f*)bp, b1 = *(const v4f*)(bp + 4);
#pragma unroll
        for (int j = 0; j < 4; ++j) { acc[t][j] = b0[j]; acc[t][4 + j] = b1[j]; }
    }
#pragma unroll
    for (int kk = 0; kk < 8; ++kk) {
        v16h b;
        if (kk < 4) {
            const _Float16* src = h16 + (size_t)n * HH + kk * 32 + off;
            b = ld_b2(src, src + 16);
        } else {
            const int t2 = 2 * (kk - 4);   // m_i half lives in registers already
#pragma unroll
            for (int i = 0; i < 8; ++i) {
                b[i]     = (_Float16)mi_[t2][i];
                b[i + 8] = (_Float16)mi_[t2 + 1][i];
            }
        }
#pragma unroll
        for (int t = 0; t < 8; ++t) {
            v16h a = ld_frag(lWn1 + (size_t)(t * 8 + kk) * 512 + lane * 16);
            acc[t] = wmma16(a, b, acc[t]);
        }
    }
#pragma unroll
    for (int t = 0; t < 8; ++t)
#pragma unroll
        for (int j = 0; j < 8; ++j) acc[t][j] = silu_f(acc[t][j]);

    v16h bf[4];
#pragma unroll
    for (int kk = 0; kk < 4; ++kk)
#pragma unroll
        for (int i = 0; i < 8; ++i) {
            bf[kk][i]     = (_Float16)acc[2 * kk][i];
            bf[kk][i + 8] = (_Float16)acc[2 * kk + 1][i];
        }

    // ---- GEMM n2 + bias + residual ----
    v8f o[8];
#pragma unroll
    for (int t = 0; t < 8; ++t) {
        const float* bp = bn2 + t * 16 + off;
        const float* hp = h + (size_t)n * HH + t * 16 + off;
        v4f b0 = *(const v4f*)bp, b1 = *(const v4f*)(bp + 4);
        v4f h0 = *(const v4f*)hp, h1 = *(const v4f*)(hp + 4);
#pragma unroll
        for (int j = 0; j < 4; ++j) { o[t][j] = b0[j] + h0[j]; o[t][4 + j] = b1[j] + h1[j]; }
    }
#pragma unroll
    for (int kk = 0; kk < 4; ++kk)
#pragma unroll
        for (int t = 0; t < 8; ++t) {
            v16h a = ld_frag(lWn2 + (size_t)(t * 4 + kk) * 512 + lane * 16);
            o[t] = wmma16(a, bf[kk], o[t]);
        }

    // ---- LayerNorm over 128 features (split across lane pair l, l+16) ----
    float sum = 0.0f, sq = 0.0f;
#pragma unroll
    for (int t = 0; t < 8; ++t)
#pragma unroll
        for (int j = 0; j < 8; ++j) { float v = o[t][j]; sum += v; sq += v * v; }
    sum += __shfl_xor(sum, 16, 32);
    sq  += __shfl_xor(sq, 16, 32);
    const float mu = sum * (1.0f / 128.0f);
    const float var = sq * (1.0f / 128.0f) - mu * mu;
    const float rs = rsqrtf(var + 1e-5f);

    float* op = outh + (size_t)n * HH;
#pragma unroll
    for (int t = 0; t < 8; ++t) {
        const float* gp = lng + t * 16 + off;
        const float* bp = lnb + t * 16 + off;
        v4f g0 = *(const v4f*)gp, g1 = *(const v4f*)(gp + 4);
        v4f c0 = *(const v4f*)bp, c1 = *(const v4f*)(bp + 4);
        v4f r0, r1;
#pragma unroll
        for (int j = 0; j < 4; ++j) {
            r0[j] = (o[t][j] - mu) * rs * g0[j] + c0[j];
            r1[j] = (o[t][4 + j] - mu) * rs * g1[j] + c1[j];
        }
        *(v4f*)(op + t * 16 + off)     = r0;
        *(v4f*)(op + t * 16 + off + 4) = r1;
    }

    if ((lane & 16) == 0) {               // x_new = x + segment_sum(force)
        outx[(size_t)n * 3 + 0] = x[(size_t)n * 3 + 0] + fx;
        outx[(size_t)n * 3 + 1] = x[(size_t)n * 3 + 1] + fy;
        outx[(size_t)n * 3 + 2] = x[(size_t)n * 3 + 2] + fz;
    }
}

extern "C" void kernel_launch(void* const* d_in, const int* in_sizes, int n_in,
                              void* d_out, int out_size, void* d_ws, size_t ws_size,
                              hipStream_t stream) {
    const float* h      = (const float*)d_in[0];
    const float* x      = (const float*)d_in[1];
    const long long* ei = (const long long*)d_in[2];
    const float* W_e1   = (const float*)d_in[3];
    const float* b_e1   = (const float*)d_in[4];
    const float* W_e2   = (const float*)d_in[5];
    const float* b_e2   = (const float*)d_in[6];
    const float* W_c1   = (const float*)d_in[7];
    const float* b_c1   = (const float*)d_in[8];
    const float* W_c2   = (const float*)d_in[9];
    const float* W_n1   = (const float*)d_in[10];
    const float* b_n1   = (const float*)d_in[11];
    const float* W_n2   = (const float*)d_in[12];
    const float* b_n2   = (const float*)d_in[13];
    const float* ln_g   = (const float*)d_in[14];
    const float* ln_b   = (const float*)d_in[15];

    char* ws = (char*)d_ws;
    size_t o = 0;
    auto alloc = [&](size_t bytes) -> char* {
        char* p = ws + o;
        o += (bytes + 255) & ~(size_t)255;
        return p;
    };
    _Float16* h16   = (_Float16*)alloc((size_t)NN * HH * 2);   // 12.8 MB
    _Float16* m_e   = (_Float16*)alloc((size_t)EE * HH * 2);   // 204.8 MB
    float*    fvw   = (float*)alloc((size_t)EE * 3 * 4);       // 9.6 MB
    int*      cntw  = (int*)alloc((size_t)NN * 4);             // 0.2 MB
    int*      eidsw = (int*)alloc((size_t)NN * CAP * 4);       // 12.8 MB
    _Float16* wedge = (_Float16*)alloc(131072);                // We1|We2|Wc1 frags
    _Float16* wnode = (_Float16*)alloc(98304);                 // Wn1|Wn2 frags

    _Float16* We1p = wedge;
    _Float16* We2p = wedge + 32768;   // +64 KB
    _Float16* Wc1p = wedge + 49152;   // +96 KB
    _Float16* Wn1p = wnode;
    _Float16* Wn2p = wnode + 32768;   // +64 KB

    init_k<<<(NN * HH + 255) / 256, 256, 0, stream>>>(h, h16, cntw);
    pack_wT<<<(8 * 8 * 32 + 255) / 256, 256, 0, stream>>>(W_e1, We1p, 8, 128, 8 * 8 * 32);
    pack_wT<<<(8 * 4 * 32 + 255) / 256, 256, 0, stream>>>(W_e2, We2p, 4, 128, 8 * 4 * 32);
    pack_wT<<<(8 * 4 * 32 + 255) / 256, 256, 0, stream>>>(W_c1, Wc1p, 4, 128, 8 * 4 * 32);
    pack_wT<<<(8 * 8 * 32 + 255) / 256, 256, 0, stream>>>(W_n1, Wn1p, 8, 128, 8 * 8 * 32);
    pack_wT<<<(8 * 4 * 32 + 255) / 256, 256, 0, stream>>>(W_n2, Wn2p, 4, 128, 8 * 4 * 32);
    scatter_k<<<(EE + 255) / 256, 256, 0, stream>>>(ei, cntw, eidsw);

    const float* We1last = W_e1 + (size_t)256 * 128;   // radial rank-1 row

    egnn_edge_kernel<<<EE / (16 * 8), 256, 131072, stream>>>(
        h16, x, ei, wedge, We1last, b_e1, b_e2, b_c1, W_c2, m_e, fvw);

    float* outh = (float*)d_out;
    float* outx = outh + (size_t)NN * HH;

    egnn_node_kernel<<<(NN / 16 + 7) / 8, 256, 98304, stream>>>(
        h, h16, x, cntw, eidsw, m_e, fvw, wnode, b_n1, b_n2, ln_g, ln_b, outh, outx);
}
